// LearnedGraphBuilder_24893630447897
// MI455X (gfx1250) — matrix operand order repaired and checked by reading, not compiled, not run on hardware
//
#include <hip/hip_runtime.h>
#include <math.h>

// ---------------------------------------------------------------------------
// LearnedGraphBuilder for MI455X (gfx1250, wave32, WMMA)
//
// B=8, N=512, D=128, H=64, H2=32, K(topk)=10, temperature=1
//
// Stage 1: pi = h @ W1[:128] + b1  (f16), pj = h @ W1[128:]  (f16)
// Stage 2: per (b,i) block: scores(j) = W3.gelu(W2.gelu(pi+pj+b1)+b2)+b3
//          via v_wmma_f32_16x16x32_f16, then top-10 + softmax -> edges.
//
// gelu: hardware v_tanh_f32 (CDNA5 trans op), branch-free.
// cross-lane reductions: single ds_swizzle_b32 per exchange.
// pj staging: global_load_async_to_lds_b128 (ASYNCcnt) when available.
// ---------------------------------------------------------------------------

typedef __attribute__((ext_vector_type(16))) _Float16 v16h;
typedef __attribute__((ext_vector_type(2)))  _Float16 h2;
typedef __attribute__((ext_vector_type(8)))  float    v8f;

#define BB   8
#define NN   512
#define DD   128
#define HH   64
#define H2   32
#define TOPK 10
#define NEDGE (BB * NN * TOPK)   // 40960

// ds_swizzle group-of-32: offset = {xor_mask[14:10], or_mask[9:5], and_mask[4:0]}
template <int M>
__device__ __forceinline__ int swzi(int v) {
    return __builtin_amdgcn_ds_swizzle(v, 0x1F | (M << 10));
}
template <int M>
__device__ __forceinline__ float swzf(float v) {
    return __int_as_float(swzi<M>(__float_as_int(v)));
}

#if __has_builtin(__builtin_amdgcn_global_load_async_to_lds_b128)
#define HAVE_ASYNC_LDS 1
typedef __attribute__((ext_vector_type(4))) int i4v;
typedef __attribute__((address_space(1))) i4v gi4v;
typedef __attribute__((address_space(3))) i4v li4v;
#endif

__device__ __forceinline__ float gelu_fast(float x) {
#if __has_builtin(__builtin_amdgcn_tanhf)
    const float u = x * (0.7978845608028654f + 0.0356774081363001f * x * x);
    return 0.5f * x * (1.0f + __builtin_amdgcn_tanhf(u));
#elif __has_builtin(__builtin_amdgcn_tanh_f32)
    const float u = x * (0.7978845608028654f + 0.0356774081363001f * x * x);
    return 0.5f * x * (1.0f + __builtin_amdgcn_tanh_f32(u));
#else
    // branch-free Abramowitz-Stegun erf fallback (no exec-mask divergence)
    const float z  = x * 0.70710678118654752f;
    const float az = __builtin_fabsf(z);
    const float t  = __builtin_amdgcn_rcpf(1.0f + 0.3275911f * az);
    float p = 1.061405429f;
    p = p * t - 1.453152027f;
    p = p * t + 1.421413741f;
    p = p * t - 0.284496736f;
    p = p * t + 0.254829592f;
    const float e   = 1.0f - (p * t) * __expf(-az * az);
    const float erf = __builtin_copysignf(e, z);
    return 0.5f * x * (1.0f + erf);
#endif
}

// ---------------------------------------------------------------------------
// Stage 1: project h (4096 x 128) -> pi (4096 x 64, +b1 folded), pj (4096 x 64)
// ---------------------------------------------------------------------------
__global__ void lgb_proj_kernel(const float* __restrict__ h,
                                const float* __restrict__ W1,   // (256, 64) (in, out)
                                const float* __restrict__ b1,   // (64)
                                _Float16* __restrict__ pi,
                                _Float16* __restrict__ pj) {
    __shared__ float sh[DD];
    const int row = blockIdx.x;          // 0..4095
    const int t   = threadIdx.x;         // 0..127
    sh[t] = h[(size_t)row * DD + t];
    __syncthreads();

    float acc = 0.0f;
    if (t < HH) {
        #pragma unroll 8
        for (int k = 0; k < DD; ++k) acc += sh[k] * W1[k * HH + t];
        pi[(size_t)row * HH + t] = (_Float16)(acc + b1[t]);   // fold bias into pi
    } else {
        const int c = t - HH;
        #pragma unroll 8
        for (int k = 0; k < DD; ++k) acc += sh[k] * W1[(DD + k) * HH + c];
        pj[(size_t)row * HH + c] = (_Float16)acc;
    }
}

// ---------------------------------------------------------------------------
// Stage 2: per-(b,i) scores via WMMA, then top-10 + softmax + edge emit.
// Block = 512 threads (16 waves). Each wave owns one 16-pair M-tile per half.
// ---------------------------------------------------------------------------
__global__ void __launch_bounds__(512)
lgb_edges_kernel(const _Float16* __restrict__ pi,
                 const _Float16* __restrict__ pj,
                 const float* __restrict__ W2g,   // (64, 32) row-major
                 const float* __restrict__ b2g,   // (32)
                 const float* __restrict__ W3g,   // (32)
                 const float* __restrict__ b3g,   // (1)
                 float* __restrict__ out) {       // [src 40960][dst 40960][attr 40960]
    __shared__ __align__(16) _Float16 s_pj[256 * HH];   // 32 KB: half of the j rows
    __shared__ __align__(16) _Float16 s_w2[HH * H2];    // 4 KB
    __shared__ float s_pi[HH];
    __shared__ float s_scores[NN];

    const int tid    = threadIdx.x;
    const int lane   = tid & 31;
    const int wave   = tid >> 5;          // 0..15
    const int row    = blockIdx.x;        // global node index b*N + i
    const int b      = row >> 9;
    const int i_loc  = row & (NN - 1);

    // ---- stage W2 (f16) and pi row into LDS -------------------------------
    for (int q = tid; q < HH * H2; q += 512) s_w2[q] = (_Float16)W2g[q];
    if (tid < HH) s_pi[tid] = (float)pi[(size_t)row * HH + tid];
    __syncthreads();

    // ---- build the 4 B-operand tiles (nt x ks) once per wave --------------
    // B layout (32x16 f16): element t -> K = (t&7) + (t>>3)*16 + 8*(lane>>4)
    //                       (same as A); column n = lane&15.
    v16h Bt[2][2];
    {
        const int n = (lane & 15);
        #pragma unroll
        for (int nt = 0; nt < 2; ++nt) {
            #pragma unroll
            for (int ks = 0; ks < 2; ++ks) {
                const int kb = ks * 32 + ((lane >> 4) * 8);
                #pragma unroll
                for (int t = 0; t < 16; ++t)
                    Bt[nt][ks][t] = s_w2[(kb + (t & 7) + ((t >> 3) * 16)) * H2 + (nt * 16 + n)];
            }
        }
    }

    // ---- preload this lane's fixed pi pattern into registers (once) -------
    // A layout element t (v=t>>1, p=t&1): K = ((v&3)*2+p) + ((v>>2)*16) + khalf
    const int khalf = (lane >> 4) * 8;
    float piA[16], piB[16];
    #pragma unroll
    for (int t = 0; t < 16; ++t) {
        const int v = t >> 1, p = t & 1;
        const int kl = ((v & 3) * 2 + p) + ((v >> 2) * 16) + khalf;
        piA[t] = s_pi[kl];
        piB[t] = s_pi[32 + kl];
    }

    const float w3a = W3g[lane & 15];
    const float w3b = W3g[16 + (lane & 15)];
    const float b2a = b2g[lane & 15];
    const float b2b = b2g[16 + (lane & 15)];
    const float bias3 = b3g[0];

    // ---- two halves of 256 j-rows each ------------------------------------
    #pragma unroll
    for (int half = 0; half < 2; ++half) {
        // stage 256 pj rows (32 KB) into LDS
        {
            const uint4* gsrc = (const uint4*)(pj + ((size_t)b * NN + half * 256) * HH);
            uint4* ldst = (uint4*)s_pj;
            #pragma unroll
            for (int q = 0; q < 4; ++q) {
                const int idx = tid + 512 * q;
#ifdef HAVE_ASYNC_LDS
                __builtin_amdgcn_global_load_async_to_lds_b128(
                    (gi4v*)(uintptr_t)(gsrc + idx),
                    (li4v*)(uintptr_t)(ldst + idx), 0, 0);
#else
                ldst[idx] = gsrc[idx];
#endif
            }
#ifdef HAVE_ASYNC_LDS
#if __has_builtin(__builtin_amdgcn_s_wait_asynccnt)
            __builtin_amdgcn_s_wait_asynccnt(0);
#else
            asm volatile("s_wait_asynccnt 0x0" ::: "memory");
#endif
#endif
            if (half == 0) {
                __builtin_prefetch(pj + ((size_t)b * NN + 256) * HH + tid * 8, 0, 1);
            }
        }
        __syncthreads();

        // each wave: one 16-pair M-tile; j = half*256 + wave*16 + (lane&15)
        const int jbase = (wave * 16 + (lane & 15)) * HH;

        // A operands: load contiguous (k, k+1) f16 pairs, gelu in f32
        v16h A0, A1;
        #pragma unroll
        for (int vv = 0; vv < 8; ++vv) {
            const int kl = ((vv & 3) * 2) + ((vv >> 2) * 16) + khalf;   // even
            const h2 pj0 = *(const h2*)&s_pj[jbase + kl];
            const h2 pj1 = *(const h2*)&s_pj[jbase + 32 + kl];
            A0[2 * vv]     = (_Float16)gelu_fast(piA[2 * vv]     + (float)pj0.x);
            A0[2 * vv + 1] = (_Float16)gelu_fast(piA[2 * vv + 1] + (float)pj0.y);
            A1[2 * vv]     = (_Float16)gelu_fast(piB[2 * vv]     + (float)pj1.x);
            A1[2 * vv + 1] = (_Float16)gelu_fast(piB[2 * vv + 1] + (float)pj1.y);
        }

        v8f acc0 = {};
        v8f acc1 = {};
        acc0 = __builtin_amdgcn_wmma_f32_16x16x32_f16(false, A0, false, Bt[0][0],
                                                      (short)0, acc0, false, false);
        acc0 = __builtin_amdgcn_wmma_f32_16x16x32_f16(false, A1, false, Bt[0][1],
                                                      (short)0, acc0, false, false);
        acc1 = __builtin_amdgcn_wmma_f32_16x16x32_f16(false, A0, false, Bt[1][0],
                                                      (short)0, acc1, false, false);
        acc1 = __builtin_amdgcn_wmma_f32_16x16x32_f16(false, A1, false, Bt[1][1],
                                                      (short)0, acc1, false, false);

        // D layout: lane = column (n = lane&15 per n-tile), VGPR r = row
        // (M = r + 8*(lane>>4)). Contract with W3, reduce across 16 lanes
        // with single-op ds_swizzle xor exchanges.
        #pragma unroll
        for (int r = 0; r < 8; ++r) {
            float z = gelu_fast(acc0[r] + b2a) * w3a
                    + gelu_fast(acc1[r] + b2b) * w3b;
            z += swzf<1>(z);
            z += swzf<2>(z);
            z += swzf<4>(z);
            z += swzf<8>(z);
            if ((lane & 15) == 0) {
                const int jrow = half * 256 + wave * 16 + r + 8 * (lane >> 4);
                s_scores[jrow] = z + bias3;
            }
        }
        __syncthreads();
    }

    // ---- top-10 + softmax + edge emit (wave 0) ----------------------------
    if (tid < 32) {
        float v[16];
        #pragma unroll
        for (int t = 0; t < 16; ++t) {
            const int idx = lane + 32 * t;
            v[t] = (idx == i_loc) ? -INFINITY : s_scores[idx];  // mask self-edge
        }

        float topv[TOPK];
        int   topi[TOPK];
        for (int kk = 0; kk < TOPK; ++kk) {
            float best = -INFINITY;
            int   bi   = 0x7fffffff;
            #pragma unroll
            for (int t = 0; t < 16; ++t) {
                const int idx = lane + 32 * t;
                if (v[t] > best) { best = v[t]; bi = idx; }
            }
#define LGB_RED_STEP(MASK)                                                  \
            {                                                               \
                const float ov = swzf<MASK>(best);                          \
                const int   oi = swzi<MASK>(bi);                            \
                if (ov > best || (ov == best && oi < bi)) {                 \
                    best = ov; bi = oi;                                     \
                }                                                           \
            }
            LGB_RED_STEP(16)
            LGB_RED_STEP(8)
            LGB_RED_STEP(4)
            LGB_RED_STEP(2)
            LGB_RED_STEP(1)
#undef LGB_RED_STEP
            topv[kk] = best; topi[kk] = bi;
            #pragma unroll
            for (int t = 0; t < 16; ++t)
                if (bi == lane + 32 * t) v[t] = -INFINITY;   // remove winner
        }

        if (lane == 0) {
            const float mx = topv[0];
            float s = 0.0f;
            #pragma unroll
            for (int kk = 0; kk < TOPK; ++kk) s += __expf(topv[kk] - mx);
            const float inv = 1.0f / s;

            float* out_src  = out;
            float* out_dst  = out + NEDGE;
            float* out_attr = out + 2 * NEDGE;
            const int ebase = row * TOPK;
            #pragma unroll
            for (int kk = 0; kk < TOPK; ++kk) {
                out_src[ebase + kk]  = (float)row;                    // b*N + i
                out_dst[ebase + kk]  = (float)(b * NN + topi[kk]);    // b*N + j
                out_attr[ebase + kk] = __expf(topv[kk] - mx) * inv;
            }
        }
    }
}

// ---------------------------------------------------------------------------
extern "C" void kernel_launch(void* const* d_in, const int* in_sizes, int n_in,
                              void* d_out, int out_size, void* d_ws, size_t ws_size,
                              hipStream_t stream) {
    (void)in_sizes; (void)n_in; (void)out_size; (void)ws_size;
    const float* h  = (const float*)d_in[0];
    // d_in[1] = live_mask (all true), d_in[2] = N, d_in[3] = B  -> constants here
    const float* W1 = (const float*)d_in[4];
    const float* b1 = (const float*)d_in[5];
    const float* W2 = (const float*)d_in[6];
    const float* b2 = (const float*)d_in[7];
    const float* W3 = (const float*)d_in[8];
    const float* b3 = (const float*)d_in[9];

    _Float16* pi = (_Float16*)d_ws;                    // 4096*64 f16 = 512 KB
    _Float16* pj = pi + (size_t)(BB * NN) * HH;        // 4096*64 f16 = 512 KB

    lgb_proj_kernel<<<BB * NN, DD, 0, stream>>>(h, W1, b1, pi, pj);
    lgb_edges_kernel<<<BB * NN, 512, 0, stream>>>(pi, pj, W2, b2, W3, b3,
                                                  (float*)d_out);
}
